// HPFModel_74320114090567
// MI455X (gfx1250) — matrix operand order repaired and checked by reading, not compiled, not run on hardware
//
#include <hip/hip_runtime.h>
#include <hip/hip_bf16.h>
#include <stdint.h>

// ---------------------------------------------------------------------------
// HPF-GNN on MI455X (gfx1250).
// Key insight: Wmat = relu(EPS*I - D^-1/2 (A+I) D^-1/2) is DIAGONAL for any
// edge list (off-diagonal entries are always <= 0), so each GCN layer is
// h' = w[i] * (h @ W) + b.  We run three fused WMMA GEMMs (f16 in, f32 acc)
// with row-scale/bias/ReLU/LayerNorm epilogues fused via LDS.
// ---------------------------------------------------------------------------

typedef __attribute__((ext_vector_type(16))) _Float16 v16h;
typedef __attribute__((ext_vector_type(8)))  float    v8f;

#define NNODES 8192
#define NEDGES 262144
#define FIN    512
#define HID    256
#define NCLS   16
#define LN_EPS 1e-5f
#define EPS_HPF 0.4f

// ---------------------------------------------------------------- utilities
__global__ void zero_i32_kernel(int* __restrict__ p, int n) {
  int i = blockIdx.x * blockDim.x + threadIdx.x;
  if (i < n) p[i] = 0;
}

__global__ void edge_hist_kernel(const int* __restrict__ ei,
                                 int* __restrict__ outdeg,
                                 int* __restrict__ selfc) {
  int e = blockIdx.x * blockDim.x + threadIdx.x;
  if (e < NEDGES) {
    int s = ei[e];            // edge_index[0, e]  (source)
    int t = ei[NEDGES + e];   // edge_index[1, e]  (target)
    atomicAdd(&outdeg[s], 1);
    if (s == t) atomicAdd(&selfc[s], 1);
  }
}

// w[i] = max(0, EPS - (gamma*self_i + alpha) / (gamma*outdeg_i + alpha))
__global__ void compute_w_kernel(const int* __restrict__ outdeg,
                                 const int* __restrict__ selfc,
                                 float* __restrict__ w) {
  int i = blockIdx.x * blockDim.x + threadIdx.x;
  if (i < NNODES) {
    float rowsum = (float)outdeg[i] + 1.0f;   // gamma=1, alpha=1
    float diag   = (float)selfc[i]  + 1.0f;
    float u = EPS_HPF - diag / rowsum;
    w[i] = u > 0.0f ? u : 0.0f;
  }
}

__global__ void f32_to_f16_kernel(const float* __restrict__ src,
                                  _Float16* __restrict__ dst, int n) {
  int i = blockIdx.x * blockDim.x + threadIdx.x;
  if (i < n) dst[i] = (_Float16)src[i];
}

// W: K x N row-major (f32)  ->  Wt: N x K row-major (f16)
__global__ void transpose_f16_kernel(const float* __restrict__ W,
                                     _Float16* __restrict__ Wt, int K, int N) {
  int idx = blockIdx.x * blockDim.x + threadIdx.x;
  if (idx < K * N) {
    int n = idx / K;
    int k = idx % K;
    Wt[idx] = (_Float16)W[(size_t)k * N + n];
  }
}

// ------------------------------------------------------- fused GEMM + LN
// Block: 512 threads = 16 waves. Wave `w` computes the 16x16 output tile at
// columns [16w, 16w+16) of row slab [16*blockIdx.x, +16). Epilogue:
// val = rowscale[m]*acc + bias[n] ; relu ; LayerNorm over all 256 cols;
// output stored as f16 (input to next layer).
__global__ __launch_bounds__(512)
void gemm16_ln_kernel(const _Float16* __restrict__ A,    // M x Kdim, f16
                      const _Float16* __restrict__ Wt,   // 256 x Kdim, f16
                      const float* __restrict__ rowscale,
                      const float* __restrict__ bias,
                      const float* __restrict__ ln_g,
                      const float* __restrict__ ln_b,
                      _Float16* __restrict__ Hout,       // M x 256, f16
                      int Kdim) {
  __shared__ float smem[16][HID];

  const int lane  = threadIdx.x & 31;
  const int wave  = threadIdx.x >> 5;   // 0..15 -> n-tile
  const int hi    = lane >> 4;          // half-wave: 0 or 1
  const int l16   = lane & 15;
  const int mBase = blockIdx.x * 16;
  const int nBase = wave * 16;
  const int m = mBase + l16;            // A-fragment row for this lane
  const int n = nBase + l16;            // B-fragment col for this lane

  v8f acc = {};
  for (int k0 = 0; k0 < Kdim; k0 += 32) {
    union { v16h v; uint32_t u[8]; } a, b;
    // A 16x32 f16 layout: lane (hi,l16), pair p -> K = k0 + hi*16 + 2*p
    const uint32_t* ap = (const uint32_t*)(A + (size_t)m * Kdim + k0);
    // B 32x16 f16 layout: lane (hi,l16), element j -> K = k0 + hi*16 + j
    const uint32_t* bp = (const uint32_t*)(Wt + (size_t)n * Kdim + k0 + hi * 16);
#pragma unroll
    for (int p = 0; p < 8; ++p) {
      int koff = ((p >> 2) << 4) + hi * 8 + ((p & 3) << 1);
      a.u[p] = ap[hi * 8 + p];
      b.u[p] = bp[p];
    }
    acc = __builtin_amdgcn_wmma_f32_16x16x32_f16(
        false, a.v, false, b.v, (short)0, acc, false, false);
  }

  // C/D layout: VGPR r, lane -> row = r + hi*8, col = l16 (within tile)
#pragma unroll
  for (int r = 0; r < 8; ++r) {
    int row = r + hi * 8;
    float v = rowscale[mBase + row] * acc[r] + bias[n];
    v = v > 0.0f ? v : 0.0f;              // ReLU before LayerNorm
    smem[row][n] = v;
  }
  __syncthreads();

  // LayerNorm: wave i handles row i (32 lanes x 8 cols each)
  {
    const int row = wave;
    float vals[8];
    float s = 0.0f, s2 = 0.0f;
#pragma unroll
    for (int t = 0; t < 8; ++t) {
      float v = smem[row][lane + 32 * t];
      vals[t] = v; s += v; s2 += v * v;
    }
#pragma unroll
    for (int msk = 16; msk >= 1; msk >>= 1) {
      s  += __shfl_xor(s,  msk, 32);
      s2 += __shfl_xor(s2, msk, 32);
    }
    float mu   = s * (1.0f / HID);
    float var  = s2 * (1.0f / HID) - mu * mu;
    float rstd = rsqrtf(var + LN_EPS);
#pragma unroll
    for (int t = 0; t < 8; ++t) {
      int c = lane + 32 * t;
      float o = (vals[t] - mu) * rstd * ln_g[c] + ln_b[c];
      Hout[(size_t)(mBase + row) * HID + c] = (_Float16)o;
    }
  }
}

// ------------------------------------------ final layer (N=16) + log_softmax
// Block: 256 threads = 8 waves; each wave computes one 16x16 tile (full N).
__global__ __launch_bounds__(256)
void gemm_cls_kernel(const _Float16* __restrict__ A,   // M x 256, f16
                     const _Float16* __restrict__ Wt,  // 16 x 256, f16
                     const float* __restrict__ rowscale,
                     const float* __restrict__ bias,
                     float* __restrict__ out) {        // [emb | log_softmax]
  const int lane  = threadIdx.x & 31;
  const int wave  = threadIdx.x >> 5;   // 0..7
  const int hi    = lane >> 4;
  const int l16   = lane & 15;
  const int mBase = (blockIdx.x * 8 + wave) * 16;
  const int m = mBase + l16;
  const int n = l16;

  v8f acc = {};
  for (int k0 = 0; k0 < HID; k0 += 32) {
    union { v16h v; uint32_t u[8]; } a, b;
    const uint32_t* ap = (const uint32_t*)(A + (size_t)m * HID + k0);
    const uint32_t* bp = (const uint32_t*)(Wt + (size_t)n * HID + k0 + hi * 16);
#pragma unroll
    for (int p = 0; p < 8; ++p) {
      int koff = ((p >> 2) << 4) + hi * 8 + ((p & 3) << 1);
      a.u[p] = ap[hi * 8 + p];
      b.u[p] = bp[p];
    }
    acc = __builtin_amdgcn_wmma_f32_16x16x32_f16(
        false, a.v, false, b.v, (short)0, acc, false, false);
  }

  // Each output row lives in one 16-lane half-wave (col = l16, VGPR r).
#pragma unroll
  for (int r = 0; r < 8; ++r) {
    int row = mBase + r + hi * 8;
    float e = rowscale[row] * acc[r] + bias[n];
    float mx = e;
#pragma unroll
    for (int msk = 8; msk >= 1; msk >>= 1)
      mx = fmaxf(mx, __shfl_xor(mx, msk, 32));
    float ex = expf(e - mx);
    float sum = ex;
#pragma unroll
    for (int msk = 8; msk >= 1; msk >>= 1)
      sum += __shfl_xor(sum, msk, 32);
    float lsm = (e - mx) - logf(sum);
    out[(size_t)row * NCLS + n] = e;
    out[(size_t)NNODES * NCLS + (size_t)row * NCLS + n] = lsm;
  }
}

// ---------------------------------------------------------------------------
extern "C" void kernel_launch(void* const* d_in, const int* in_sizes, int n_in,
                              void* d_out, int out_size, void* d_ws, size_t ws_size,
                              hipStream_t stream) {
  (void)in_sizes; (void)n_in; (void)out_size; (void)ws_size;

  const float* x    = (const float*)d_in[0];
  const int*   ei   = (const int*)  d_in[1];
  const float* W0   = (const float*)d_in[2];
  const float* b0   = (const float*)d_in[3];
  const float* ln0g = (const float*)d_in[4];
  const float* ln0b = (const float*)d_in[5];
  const float* W1   = (const float*)d_in[6];
  const float* b1   = (const float*)d_in[7];
  const float* ln1g = (const float*)d_in[8];
  const float* ln1b = (const float*)d_in[9];
  const float* W2   = (const float*)d_in[10];
  const float* b2   = (const float*)d_in[11];
  float* out = (float*)d_out;

  // workspace layout (256B aligned)
  char* ws = (char*)d_ws;
  size_t off = 0;
  auto alloc = [&](size_t bytes) { void* p = ws + off; off = (off + bytes + 255) & ~(size_t)255; return p; };
  _Float16* Xh  = (_Float16*)alloc((size_t)NNODES * FIN * 2);   // 8 MB
  _Float16* W0t = (_Float16*)alloc((size_t)HID * FIN * 2);      // 256 KB
  _Float16* W1t = (_Float16*)alloc((size_t)HID * HID * 2);      // 128 KB
  _Float16* W2t = (_Float16*)alloc((size_t)NCLS * HID * 2);     // 8 KB
  _Float16* H0  = (_Float16*)alloc((size_t)NNODES * HID * 2);   // 4 MB
  _Float16* H1  = (_Float16*)alloc((size_t)NNODES * HID * 2);   // 4 MB
  int*   outdeg = (int*)alloc((size_t)NNODES * 4);
  int*   selfc  = (int*)alloc((size_t)NNODES * 4);
  float* wdiag  = (float*)alloc((size_t)NNODES * 4);

  // 1) diagonal Wmat: histogram + closed form
  zero_i32_kernel<<<(2 * NNODES + 255) / 256, 256, 0, stream>>>(outdeg, 2 * NNODES);
  edge_hist_kernel<<<NEDGES / 256, 256, 0, stream>>>(ei, outdeg, selfc);
  compute_w_kernel<<<NNODES / 256, 256, 0, stream>>>(outdeg, selfc, wdiag);

  // 2) precision prep: X -> f16, weights -> f16 transposed (N x K, K-contig)
  f32_to_f16_kernel<<<(NNODES * FIN + 255) / 256, 256, 0, stream>>>(x, Xh, NNODES * FIN);
  transpose_f16_kernel<<<(FIN * HID + 255) / 256, 256, 0, stream>>>(W0, W0t, FIN, HID);
  transpose_f16_kernel<<<(HID * HID + 255) / 256, 256, 0, stream>>>(W1, W1t, HID, HID);
  transpose_f16_kernel<<<(HID * NCLS + 255) / 256, 256, 0, stream>>>(W2, W2t, HID, NCLS);

  // 3) layer 0: H0 = LN(relu(w * (X @ W0) + b0))
  gemm16_ln_kernel<<<NNODES / 16, 512, 0, stream>>>(Xh, W0t, wdiag, b0, ln0g, ln0b, H0, FIN);
  // 4) layer 1: H1 = LN(relu(w * (H0 @ W1) + b1))
  gemm16_ln_kernel<<<NNODES / 16, 512, 0, stream>>>(H0, W1t, wdiag, b1, ln1g, ln1b, H1, HID);
  // 5) layer 2: emb = w * (H1 @ W2) + b2 ; out = [emb | log_softmax(emb)]
  gemm_cls_kernel<<<NNODES / 128, 256, 0, stream>>>(H1, W2t, wdiag, b2, out);
}
